// GNN_22093311771370
// MI455X (gfx1250) — matrix operand order repaired and checked by reading, not compiled, not run on hardware
//
#include <hip/hip_runtime.h>

// ---------------------------------------------------------------------------
// Problem constants (match reference)
// ---------------------------------------------------------------------------
constexpr int NN   = 100000;   // nodes
constexpr int NE   = 1200000;  // edges
constexpr int FIN  = 128;      // raw features
constexpr int HID  = 256;      // MLP hidden
constexpr int DIM  = 64;       // embedding
constexpr int NCLS = 40;       // classes
constexpr int RHOP = 3;

constexpr int TILES_M = (NN + 15) / 16;   // 6250

// Packed-weight fragment sizes (bf16 elements): (K/32) * (NcPad/16) * 512
constexpr size_t PW0 = (size_t)(FIN / 32) * (HID / 16) * 512;   // 32768
constexpr size_t PW1 = (size_t)(HID / 32) * (DIM / 16) * 512;   // 16384
constexpr size_t PG  = (size_t)(DIM / 32) * (DIM / 16) * 512;   // 4096 (per hop)
constexpr size_t PD0 = (size_t)(DIM / 32) * (HID / 16) * 512;   // 16384
constexpr size_t PD1 = (size_t)(HID / 32) * (48  / 16) * 512;   // 12288 (Nc=40 padded->48)
constexpr size_t PACK_ELEMS = PW0 + PW1 + RHOP * PG + PD0 + PD1; // 90112
constexpr size_t PACK_BYTES = PACK_ELEMS * 2;                    // 180224 (256B aligned)

// ---------------------------------------------------------------------------
// WMMA types (gfx1250, wave32)
// ---------------------------------------------------------------------------
typedef __bf16 v16bf __attribute__((ext_vector_type(16)));
typedef float  v8f   __attribute__((ext_vector_type(8)));

// ---------------------------------------------------------------------------
// Pack fp32 row-major weights B[K][Nc] into WMMA B-fragment layout (bf16):
//   frag[((s*tilesN + tn)*32 + lane)*16 + i] = B[s*32 + (lane>>4)*16 + i][tn*16 + (lane&15)]
// One thread per (s, tn, lane): writes 16 contiguous bf16 (32 bytes).
// ---------------------------------------------------------------------------
__global__ __launch_bounds__(256)
void pack_b_kernel(const float* __restrict__ B, __bf16* __restrict__ out,
                   int K, int Nc, int tilesN) {
    int t = blockIdx.x * blockDim.x + threadIdx.x;
    int kSteps = K >> 5;
    int total = kSteps * tilesN * 32;
    if (t >= total) return;
    int lane  = t & 31;
    int tn    = (t >> 5) % tilesN;
    int s     = (t >> 5) / tilesN;
    int phase = lane >> 4;
    int col   = tn * 16 + (lane & 15);
    int kbase = (s << 5) + (phase << 4);
    __bf16* dst = out + (size_t)t * 16;
#pragma unroll
    for (int i = 0; i < 16; ++i) {
        float v = (col < Nc) ? B[(size_t)(kbase + i) * Nc + col] : 0.0f;
        dst[i] = (__bf16)v;
    }
}

// ---------------------------------------------------------------------------
// bf16 WMMA GEMM with N register-blocking:
//   each wave owns a 16 x (16*NT) output strip; A fragment is loaded ONCE per
//   k-step and fed to NT back-to-back v_wmma_f32_16x16x32_bf16 against NT
//   pre-swizzled B fragments (kills the tilesN-fold redundant A traffic).
// ---------------------------------------------------------------------------
template <int NT, bool HAS_BIAS, bool DO_RELU>
__global__ __launch_bounds__(256)
void wmma_gemm_kernel(const float* __restrict__ A, const __bf16* __restrict__ Bp,
                      const float* __restrict__ bias, float* __restrict__ Cout,
                      int M, int K, int Nc, int tilesN) {
    const int lane = threadIdx.x & 31;
    const int wave = (blockIdx.x * blockDim.x + threadIdx.x) >> 5;
    const int strips = tilesN / NT;
    const int tilesM = (M + 15) >> 4;
    if (wave >= tilesM * strips) return;          // wave-uniform: EXEC stays all-ones
    const int tm    = wave / strips;
    const int tn0   = (wave - tm * strips) * NT;  // first N tile of this strip
    const int phase = lane >> 4;

    // A-fragment row for this lane (clamp OOB rows; they only feed masked stores)
    const int mrow = tm * 16 + (lane & 15);
    const int arow_i = (mrow < M) ? mrow : (M - 1);
    const float* __restrict__ arow = A + (size_t)arow_i * K;

    const int kSteps = K >> 5;
    // B fragment base for tile (s, tn): Bp + ((s*tilesN + tn)*32 + lane)*16
    const __bf16* __restrict__ bbase = Bp + ((size_t)tn0 * 32 + lane) * 16;
    const size_t bstrideS = (size_t)tilesN * 512;  // elems per k-step
    constexpr size_t bstrideT = 512;               // elems per N tile

    v8f acc[NT];
#pragma unroll
    for (int j = 0; j < NT; ++j) acc[j] = v8f{};

    for (int s = 0; s < kSteps; ++s) {
        // 16-bit A 16x32 layout (matches packed B): elems 0..15 = K[k0..k0+15], k0 = s*32 + phase*16
        const int k0 = (s << 5) + (phase << 4);
        float4 a0 = *(const float4*)(arow + k0);
        float4 a1 = *(const float4*)(arow + k0 + 4);
        float4 a2 = *(const float4*)(arow + k0 + 8);
        float4 a3 = *(const float4*)(arow + k0 + 12);
        v16bf af;
        af[0]  = (__bf16)a0.x; af[1]  = (__bf16)a0.y; af[2]  = (__bf16)a0.z; af[3]  = (__bf16)a0.w;
        af[4]  = (__bf16)a1.x; af[5]  = (__bf16)a1.y; af[6]  = (__bf16)a1.z; af[7]  = (__bf16)a1.w;
        af[8]  = (__bf16)a2.x; af[9]  = (__bf16)a2.y; af[10] = (__bf16)a2.z; af[11] = (__bf16)a2.w;
        af[12] = (__bf16)a3.x; af[13] = (__bf16)a3.y; af[14] = (__bf16)a3.z; af[15] = (__bf16)a3.w;

        const __bf16* bs = bbase + (size_t)s * bstrideS;
#pragma unroll
        for (int j = 0; j < NT; ++j) {
            v16bf bf_ = *(const v16bf*)(bs + (size_t)j * bstrideT);  // 2x coalesced b128
            acc[j] = __builtin_amdgcn_wmma_f32_16x16x32_bf16(
                /*neg_a=*/false, af, /*neg_b=*/false, bf_,
                /*c_mod=*/(short)0, acc[j], /*reuse_a=*/false, /*reuse_b=*/false);
        }
    }

    // C/D layout: VGPR v -> M = phase*8 + v, N = lane&15
#pragma unroll
    for (int j = 0; j < NT; ++j) {
        const int col = (tn0 + j) * 16 + (lane & 15);
        if (col >= Nc) continue;
        float bv = 0.0f;
        if (HAS_BIAS) bv = bias[col];
#pragma unroll
        for (int v = 0; v < 8; ++v) {
            int r = tm * 16 + (phase << 3) + v;
            if (r < M) {
                float val = acc[j][v] + bv;
                if (DO_RELU) val = val > 0.0f ? val : 0.0f;
                Cout[(size_t)r * Nc + col] = val;
            }
        }
    }
}

// ---------------------------------------------------------------------------
// Edge scatter-add: h[dst[e]][d] += t[src[e]][d]  (thread per (edge, feature))
// ---------------------------------------------------------------------------
__global__ __launch_bounds__(256)
void scatter_add_kernel(const float* __restrict__ T, const int* __restrict__ src,
                        const int* __restrict__ dst, float* __restrict__ H,
                        long long total) {
    long long t = (long long)blockIdx.x * blockDim.x + threadIdx.x;
    if (t >= total) return;
    int d = (int)(t & 63);
    long long e = t >> 6;
    int s  = src[e];
    int dd = dst[e];
    atomicAdd(&H[(size_t)dd * 64 + d], T[(size_t)s * 64 + d]);
}

__global__ __launch_bounds__(256)
void fill_zero_kernel(float* __restrict__ p, long long n) {
    long long t = (long long)blockIdx.x * blockDim.x + threadIdx.x;
    if (t < n) p[t] = 0.0f;
}

__global__ __launch_bounds__(256)
void bias_relu_kernel(float* __restrict__ h, const float* __restrict__ b, long long n) {
    long long t = (long long)blockIdx.x * blockDim.x + threadIdx.x;
    if (t >= n) return;
    float v = h[t] + b[t & 63];
    h[t] = v > 0.0f ? v : 0.0f;
}

__global__ __launch_bounds__(256)
void log_softmax_kernel(const float* __restrict__ logits, float* __restrict__ out,
                        int N, int C) {
    int r = blockIdx.x * blockDim.x + threadIdx.x;
    if (r >= N) return;
    const float* row = logits + (size_t)r * C;
    float m = row[0];
    for (int i = 1; i < C; ++i) { float v = row[i]; m = v > m ? v : m; }
    float sum = 0.0f;
    for (int i = 0; i < C; ++i) sum += __expf(row[i] - m);
    float lse = m + __logf(sum);
    float* orow = out + (size_t)r * C;
    for (int i = 0; i < C; ++i) orow[i] = row[i] - lse;
}

// ---------------------------------------------------------------------------
// Host driver
// ---------------------------------------------------------------------------
extern "C" void kernel_launch(void* const* d_in, const int* in_sizes, int n_in,
                              void* d_out, int out_size, void* d_ws, size_t ws_size,
                              hipStream_t stream) {
    const float* x      = (const float*)d_in[0];
    const int*   eidx   = (const int*)d_in[1];
    const float* enc_w0 = (const float*)d_in[2];
    const float* enc_b0 = (const float*)d_in[3];
    const float* enc_w1 = (const float*)d_in[4];
    const float* enc_b1 = (const float*)d_in[5];
    const float* gcn_w  = (const float*)d_in[6];
    const float* gcn_b  = (const float*)d_in[7];
    const float* dec_w0 = (const float*)d_in[8];
    const float* dec_b0 = (const float*)d_in[9];
    const float* dec_w1 = (const float*)d_in[10];
    const float* dec_b1 = (const float*)d_in[11];
    float* out = (float*)d_out;

    // Workspace layout
    __bf16* pk  = (__bf16*)d_ws;
    __bf16* pw0 = pk;
    __bf16* pw1 = pw0 + PW0;
    __bf16* pg0 = pw1 + PW1;
    __bf16* pd0 = pg0 + RHOP * PG;
    __bf16* pd1 = pd0 + PD0;
    float* fbase = (float*)((char*)d_ws + PACK_BYTES);
    float* H1 = fbase;                       // [NN, 256]  (encoder hidden / decoder hidden)
    float* Hb = H1 + (size_t)NN * HID;       // [NN, 64]   node embeddings
    float* Tb = Hb + (size_t)NN * DIM;       // [NN, 64]   transformed messages
    float* lg = Tb + (size_t)NN * DIM;       // [NN, 40]   logits

    // ---- pack all weights into bf16 WMMA B fragments ----
    auto packLaunch = [&](const float* B, __bf16* o, int K, int Nc, int NcPad) {
        int tilesN = NcPad / 16;
        int threads = (K / 32) * tilesN * 32;
        pack_b_kernel<<<(threads + 255) / 256, 256, 0, stream>>>(B, o, K, Nc, tilesN);
    };
    packLaunch(enc_w0, pw0, FIN, HID, HID);
    packLaunch(enc_w1, pw1, HID, DIM, DIM);
    for (int r = 0; r < RHOP; ++r)
        packLaunch(gcn_w + (size_t)r * DIM * DIM, pg0 + (size_t)r * PG, DIM, DIM, DIM);
    packLaunch(dec_w0, pd0, DIM, HID, HID);
    packLaunch(dec_w1, pd1, HID, NCLS, 48);

    auto gemmBlocks = [&](int tilesN, int nt) {
        long long waves = (long long)TILES_M * (tilesN / nt);
        return (unsigned)((waves * 32 + 255) / 256);
    };

    // ---- encoder MLP: relu(x@W0+b0) @ W1 + b1 ----
    wmma_gemm_kernel<4, true, true ><<<gemmBlocks(HID / 16, 4), 256, 0, stream>>>(
        x,  pw0, enc_b0, H1, NN, FIN, HID, HID / 16);
    wmma_gemm_kernel<4, true, false><<<gemmBlocks(DIM / 16, 4), 256, 0, stream>>>(
        H1, pw1, enc_b1, Hb, NN, HID, DIM, DIM / 16);

    // ---- R GCN hops: transform, gather-src, scatter-add-dst, +bias, relu ----
    const int* srcI = eidx;
    const int* dstI = eidx + NE;
    const long long nElem = (long long)NN * DIM;
    const long long nScat = (long long)NE * DIM;
    for (int r = 0; r < RHOP; ++r) {
        wmma_gemm_kernel<4, false, false><<<gemmBlocks(DIM / 16, 4), 256, 0, stream>>>(
            Hb, pg0 + (size_t)r * PG, nullptr, Tb, NN, DIM, DIM, DIM / 16);
        fill_zero_kernel<<<(unsigned)((nElem + 255) / 256), 256, 0, stream>>>(Hb, nElem);
        scatter_add_kernel<<<(unsigned)((nScat + 255) / 256), 256, 0, stream>>>(
            Tb, srcI, dstI, Hb, nScat);
        bias_relu_kernel<<<(unsigned)((nElem + 255) / 256), 256, 0, stream>>>(
            Hb, gcn_b + (size_t)r * DIM, nElem);
    }

    // ---- decoder MLP + log_softmax ----
    wmma_gemm_kernel<4, true, true ><<<gemmBlocks(HID / 16, 4), 256, 0, stream>>>(
        Hb, pd0, dec_b0, H1, NN, DIM, HID, HID / 16);
    wmma_gemm_kernel<3, true, false><<<gemmBlocks(48 / 16, 3), 256, 0, stream>>>(
        H1, pd1, dec_b1, lg, NN, HID, NCLS, 48 / 16);
    log_softmax_kernel<<<(NN + 255) / 256, 256, 0, stream>>>(lg, out, NN, NCLS);
}
